// TripleCrossPaiNN_40475771797594
// MI455X (gfx1250) — matrix operand order repaired and checked by reading, not compiled, not run on hardware
//
#include <hip/hip_runtime.h>
#include <hip/hip_bf16.h>
#include <math.h>

#define Ff 64
#define NLAY 3
#define NNODE 10000
#define NEDGE 320000
#define RMAXF 5.0f
#define PIF 3.14159265358979f

typedef __attribute__((ext_vector_type(16))) _Float16 v16h;
typedef __attribute__((ext_vector_type(8)))  _Float16 v8h;
typedef __attribute__((ext_vector_type(8)))  float    v8f;
typedef unsigned int uint32x4 __attribute__((ext_vector_type(4)));
typedef int          int32x8  __attribute__((ext_vector_type(8)));
typedef int          int32x4  __attribute__((ext_vector_type(4)));

#if __has_builtin(__builtin_amdgcn_tensor_load_to_lds) && \
    __has_builtin(__builtin_amdgcn_s_wait_tensorcnt)
#define USE_TDM 1
#endif

// ---------------------------------------------------------------------------
// WMMA helpers (CDNA5 wave32, V_WMMA_F32_16X16X32_F16)
// ---------------------------------------------------------------------------
__device__ __forceinline__ v8f wmma16(v16h a, v16h b, v8f c) {
  return __builtin_amdgcn_wmma_f32_16x16x32_f16(false, a, false, b, (short)0, c,
                                                false, false);
}

// A tile (16x32 f16) from a row-major f16 matrix. Per ISA layout:
// lane L: row = L&15 ; half h = L>>4 ; VGPR0-3 hold K=(0..7)+8h, VGPR4-7 K=(16..23)+8h
// -> two contiguous 8-half loads per lane.
__device__ __forceinline__ v16h load_a16(const _Float16* A, int lda, int row0, int kt) {
  int lane = threadIdx.x & 31;
  int m = row0 + (lane & 15);
  int h = lane >> 4;
  const _Float16* p = A + (size_t)m * lda + kt * 32 + h * 8;
  v16h a;
  ((v8h*)&a)[0] = *(const v8h*)(p);
  ((v8h*)&a)[1] = *(const v8h*)(p + 16);
  return a;
}

// B tile (32x16 f16) from a pre-packed buffer laid out [nt][kt][lane][16 halves].
__device__ __forceinline__ v16h load_b16(const _Float16* Bp, int KT, int nt, int kt) {
  int lane = threadIdx.x & 31;
  return *(const v16h*)(Bp + ((((size_t)nt * KT + kt) * 32 + lane) << 4));
}

__device__ __forceinline__ float siluf(float x) { return x / (1.0f + __expf(-x)); }

// ---------------------------------------------------------------------------
// Pack f32 row-major [K][N] weight into f16 WMMA-B layout (K padded to mult of 32)
// dst index = (((nt*KT + kt)*32 + lane)*16 + j); k = kt*32 + (j<8 ? (lane>>4)*8 + j : 16 + (lane>>4)*8 + j-8)
// ---------------------------------------------------------------------------
__global__ void pack_b_kernel(const float* __restrict__ src, _Float16* __restrict__ dst,
                              int K, int N) {
  int KT = (K + 31) / 32;
  int total = (N / 16) * KT * 512;
  int idx = blockIdx.x * blockDim.x + threadIdx.x;
  if (idx >= total) return;
  int j    = idx & 15;
  int lane = (idx >> 4) & 31;
  int rest = idx >> 9;
  int kt   = rest % KT;
  int nt   = rest / KT;
  int k = kt * 32 + (j < 8 ? (lane >> 4) * 8 + j : 16 + (lane >> 4) * 8 + (j - 8));
  int n = nt * 16 + (lane & 15);
  float v = (k < K) ? src[(size_t)k * N + n] : 0.0f;
  dst[idx] = (_Float16)v;
}

// ---------------------------------------------------------------------------
// Per-edge geometry for 3 geometries: unit vec, Bessel RBF (f16, padded to 32), cutoff
// ---------------------------------------------------------------------------
__global__ void geom_kernel(const float* __restrict__ rp, const float* __restrict__ pp,
                            const float* __restrict__ ip, const int* __restrict__ ei,
                            _Float16* __restrict__ rb16, float* __restrict__ unit,
                            float* __restrict__ fc) {
  int i = blockIdx.x * blockDim.x + threadIdx.x;
  if (i >= 3 * NEDGE) return;
  int g = i / NEDGE, e = i - g * NEDGE;
  const float* pos = (g == 0) ? rp : (g == 1) ? pp : ip;
  int snd = ei[e], rcv = ei[NEDGE + e];
  float vx = pos[rcv * 3 + 0] - pos[snd * 3 + 0];
  float vy = pos[rcv * 3 + 1] - pos[snd * 3 + 1];
  float vz = pos[rcv * 3 + 2] - pos[snd * 3 + 2];
  float len = sqrtf(vx * vx + vy * vy + vz * vz);
  float inv = 1.0f / fmaxf(len, 1e-9f);
  size_t ge = (size_t)g * NEDGE + e;
  unit[ge * 3 + 0] = vx * inv;
  unit[ge * 3 + 1] = vy * inv;
  unit[ge * 3 + 2] = vz * inv;
  float lc = fminf(len, RMAXF);                 // R_OFF == 0
  float invl = 1.0f / fmaxf(lc, 1e-9f);
  float pre = sqrtf(2.0f / RMAXF);
  _Float16* rbp = rb16 + ge * 32;
  #pragma unroll 4
  for (int k = 0; k < 20; ++k)
    rbp[k] = (_Float16)(pre * sinf((float)(k + 1) * PIF * lc / RMAXF) * invl);
  #pragma unroll
  for (int k = 20; k < 32; ++k) rbp[k] = (_Float16)0.0f;
  fc[ge] = 0.5f * (cosf(PIF * lc / RMAXF) + 1.0f);
}

// ---------------------------------------------------------------------------
// Node embedding: s = atom_emb[Z] + [sin,cos] Fourier(time); v = (pp-rp) * linear_v_W
// ---------------------------------------------------------------------------
__global__ void embed_kernel(const float* __restrict__ rp, const float* __restrict__ pp,
                             const float* __restrict__ timev, const float* __restrict__ emb,
                             const float* __restrict__ fW, const float* __restrict__ lvW,
                             const int* __restrict__ Z, float* __restrict__ s,
                             float* __restrict__ v, _Float16* __restrict__ s16,
                             _Float16* __restrict__ v16b) {
  int i = blockIdx.x * blockDim.x + threadIdx.x;
  if (i >= NNODE * Ff) return;
  int n = i >> 6, f = i & 63;
  float t = timev[n];
  float proj = 2.0f * PIF * t * fW[f & 31];
  float sv = (f < 32) ? sinf(proj) : cosf(proj);
  sv += emb[(size_t)Z[n] * Ff + f];
  s[i] = sv;
  s16[i] = (_Float16)sv;
  float lw = lvW[f];
  #pragma unroll
  for (int c = 0; c < 3; ++c) {
    float d = pp[n * 3 + c] - rp[n * 3 + c];
    float vv = d * lw;
    size_t idx = ((size_t)n * 3 + c) * Ff + f;
    v[idx] = vv;
    v16b[idx] = (_Float16)vv;
  }
}

// ---------------------------------------------------------------------------
// phi = silu(s @ W1 + b1) @ W2 + b2   -> [N,192] f32. One wave per 16 nodes.
// ---------------------------------------------------------------------------
__global__ __launch_bounds__(32) void phi_kernel(
    const _Float16* __restrict__ s16, const _Float16* __restrict__ B1,
    const float* __restrict__ b1, const _Float16* __restrict__ B2,
    const float* __restrict__ b2, float* __restrict__ phi) {
  __shared__ _Float16 Hs[16 * 64];
  int m0 = blockIdx.x * 16;
  int lane = threadIdx.x & 31;
  int h = lane >> 4, nl = lane & 15;
  v16h a0 = load_a16(s16, 64, m0, 0);
  v16h a1 = load_a16(s16, 64, m0, 1);
  #pragma unroll
  for (int nt = 0; nt < 4; ++nt) {
    v8f acc = {};
    acc = wmma16(a0, load_b16(B1, 2, nt, 0), acc);
    acc = wmma16(a1, load_b16(B1, 2, nt, 1), acc);
    float bias = b1[nt * 16 + nl];
    #pragma unroll
    for (int j = 0; j < 8; ++j)
      Hs[(h * 8 + j) * 64 + nt * 16 + nl] = (_Float16)siluf(acc[j] + bias);
  }
  __syncthreads();
  v16h h0 = load_a16(Hs, 64, 0, 0);
  v16h h1 = load_a16(Hs, 64, 0, 1);
  #pragma unroll
  for (int nt = 0; nt < 12; ++nt) {
    v8f acc = {};
    acc = wmma16(h0, load_b16(B2, 2, nt, 0), acc);
    acc = wmma16(h1, load_b16(B2, 2, nt, 1), acc);
    float bias = b2[nt * 16 + nl];
    #pragma unroll
    for (int j = 0; j < 8; ++j)
      phi[(size_t)(m0 + h * 8 + j) * 192 + nt * 16 + nl] = acc[j] + bias;
  }
}

// ---------------------------------------------------------------------------
// Edge message kernel. One wave handles 16 edges; packed filter weights (36 KB,
// shared by all 8 waves of the block) are staged global->LDS with the Tensor
// Data Mover, then each geometry does 12 WMMAs; apply (.+bias)*fc, multiply by
// phi[snd], accumulate ds / dvv / unit*dvs; scatter with f32 atomics.
// ---------------------------------------------------------------------------
__global__ __launch_bounds__(256) void edge_kernel(
    const _Float16* __restrict__ rb16, const float* __restrict__ unit,
    const float* __restrict__ fcbuf, const float* __restrict__ phi,
    const float* __restrict__ v, const int* __restrict__ ei,
    const _Float16* __restrict__ Bfilt, const float* __restrict__ filt_b,
    float* __restrict__ s_acc, float* __restrict__ v_acc) {
  __shared__ __attribute__((aligned(32))) _Float16 Blds[3 * 12 * 512];  // 36 KB

#ifdef USE_TDM
  if (threadIdx.x < 32) {
    // Tensor DMA descriptor: 1-D tile, data_size=8B, 4608 qwords, LDS offset 0.
    unsigned long long ga = (unsigned long long)(const void*)Bfilt;
    const unsigned T = (3 * 12 * 512 * 2) / 8;  // 4608 qwords
    uint32x4 g0;
    g0[0] = 1u;                                        // count=1 (valid), user mode
    g0[1] = 0u;                                        // lds_addr = 0 (Blds)
    g0[2] = (unsigned)(ga & 0xFFFFFFFFu);              // global_addr[95:64]
    g0[3] = (unsigned)((ga >> 32) & 0x1FFFFFFu) | (2u << 30);  // addr hi | type=2
    int32x8 g1;
    g1[0] = (int)(3u << 16);               // wg_mask=0, data_size=3 (8B), no flags
    g1[1] = (int)((T & 0xFFFFu) << 16);    // tensor_dim0 lo16 @ bits 63:48
    g1[2] = (int)((T >> 16) | (1u << 16)); // tensor_dim0 hi16 | tensor_dim1=1 lo16
    g1[3] = (int)(T << 16);                // tensor_dim1 hi16=0 | tile_dim0=T
    g1[4] = 1;                             // tile_dim1=1, tile_dim2=0
    g1[5] = (int)T;                        // tensor_dim0_stride lo32
    g1[6] = 0;                             // stride hi16 | tensor_dim1_stride lo16
    g1[7] = 0;
    int32x4 gz = {0, 0, 0, 0};
#if __clang_major__ >= 23
    int32x8 gz8 = {0, 0, 0, 0, 0, 0, 0, 0};
    __builtin_amdgcn_tensor_load_to_lds(g0, g1, gz, gz, gz8, 0);
#else
    __builtin_amdgcn_tensor_load_to_lds(g0, g1, gz, gz, 0);
#endif
    __builtin_amdgcn_s_wait_tensorcnt(0);
  }
  __syncthreads();
#else
  for (int i = threadIdx.x; i < (3 * 12 * 512) / 8; i += 256)
    ((v8h*)Blds)[i] = ((const v8h*)Bfilt)[i];
  __syncthreads();
#endif

  int wave = threadIdx.x >> 5;
  int lane = threadIdx.x & 31;
  int tile = blockIdx.x * 8 + wave;
  int e0 = tile * 16;
  int h = lane >> 4, nl = lane & 15;

  int sndr[8], rcvr[8];
  #pragma unroll
  for (int j = 0; j < 8; ++j) {
    int e = e0 + h * 8 + j;
    sndr[j] = ei[e];
    rcvr[j] = ei[NEDGE + e];
  }

  v8f zero = {};
  v8f dsA[4], dvvA[4], uds0[4], uds1[4], uds2[4];
  #pragma unroll
  for (int t = 0; t < 4; ++t) {
    dsA[t] = zero; dvvA[t] = zero; uds0[t] = zero; uds1[t] = zero; uds2[t] = zero;
  }

  for (int g = 0; g < 3; ++g) {
    v16h a = load_a16(rb16 + (size_t)g * NEDGE * 32, 32, e0, 0);
    float fcl[8], ux[8], uy[8], uz[8];
    #pragma unroll
    for (int j = 0; j < 8; ++j) {
      size_t ge = (size_t)g * NEDGE + e0 + h * 8 + j;
      fcl[j] = fcbuf[ge];
      ux[j] = unit[ge * 3 + 0];
      uy[j] = unit[ge * 3 + 1];
      uz[j] = unit[ge * 3 + 2];
    }
    const _Float16* Bg = Blds + (size_t)g * 12 * 512;
    const float* bg = filt_b + g * 192;
    #pragma unroll
    for (int nt = 0; nt < 12; ++nt) {
      v8f acc = {};
      acc = wmma16(a, load_b16(Bg, 1, nt, 0), acc);
      float bias = bg[nt * 16 + nl];
      #pragma unroll
      for (int j = 0; j < 8; ++j) {
        float w = (acc[j] + bias) * fcl[j];
        float xj = phi[(size_t)sndr[j] * 192 + nt * 16 + nl];
        float p = w * xj;
        if (nt < 4)       dsA[nt][j] += p;
        else if (nt < 8)  dvvA[nt - 4][j] += p;
        else {
          uds0[nt - 8][j] += ux[j] * p;
          uds1[nt - 8][j] += uy[j] * p;
          uds2[nt - 8][j] += uz[j] * p;
        }
      }
    }
  }

  #pragma unroll
  for (int nt = 0; nt < 4; ++nt) {
    int col = nt * 16 + nl;
    #pragma unroll
    for (int j = 0; j < 8; ++j) {
      int sn = sndr[j], rc = rcvr[j];
      atomicAdd(&s_acc[(size_t)rc * Ff + col], dsA[nt][j]);
      float vj0 = v[((size_t)sn * 3 + 0) * Ff + col];
      float vj1 = v[((size_t)sn * 3 + 1) * Ff + col];
      float vj2 = v[((size_t)sn * 3 + 2) * Ff + col];
      atomicAdd(&v_acc[((size_t)rc * 3 + 0) * Ff + col], vj0 * dvvA[nt][j] + uds0[nt][j]);
      atomicAdd(&v_acc[((size_t)rc * 3 + 1) * Ff + col], vj1 * dvvA[nt][j] + uds1[nt][j]);
      atomicAdd(&v_acc[((size_t)rc * 3 + 2) * Ff + col], vj2 * dvvA[nt][j] + uds2[nt][j]);
    }
  }
}

// s += s_acc ; v += v_acc ; refresh f16 copies
__global__ void add_acc_kernel(float* __restrict__ s, const float* __restrict__ s_acc,
                               float* __restrict__ v, const float* __restrict__ v_acc,
                               _Float16* __restrict__ s16, _Float16* __restrict__ v16b) {
  int i = blockIdx.x * blockDim.x + threadIdx.x;
  if (i >= NNODE * Ff) return;
  int n = i >> 6, f = i & 63;
  float sn = s[i] + s_acc[i];
  s[i] = sn;
  s16[i] = (_Float16)sn;
  #pragma unroll
  for (int c = 0; c < 3; ++c) {
    size_t idx = ((size_t)n * 3 + c) * Ff + f;
    float vv = v[idx] + v_acc[idx];
    v[idx] = vv;
    v16b[idx] = (_Float16)vv;
  }
}

// Uv = v @ U ; Vv = v @ V  (rows = 3N)
__global__ __launch_bounds__(32) void uvv_kernel(const _Float16* __restrict__ v16b,
                                                 const _Float16* __restrict__ BU,
                                                 const _Float16* __restrict__ BV,
                                                 float* __restrict__ Uv,
                                                 float* __restrict__ Vv) {
  int m0 = blockIdx.x * 16;
  int lane = threadIdx.x & 31;
  int h = lane >> 4, nl = lane & 15;
  v16h a0 = load_a16(v16b, 64, m0, 0);
  v16h a1 = load_a16(v16b, 64, m0, 1);
  #pragma unroll
  for (int nt = 0; nt < 4; ++nt) {
    v8f accu = {}, accv = {};
    accu = wmma16(a0, load_b16(BU, 2, nt, 0), accu);
    accu = wmma16(a1, load_b16(BU, 2, nt, 1), accu);
    accv = wmma16(a0, load_b16(BV, 2, nt, 0), accv);
    accv = wmma16(a1, load_b16(BV, 2, nt, 1), accv);
    #pragma unroll
    for (int j = 0; j < 8; ++j) {
      size_t idx = (size_t)(m0 + h * 8 + j) * Ff + nt * 16 + nl;
      Uv[idx] = accu[j];
      Vv[idx] = accv[j];
    }
  }
}

// Vn = ||Vv||_c ; cat16 = [s16, Vn] f16 [N][128]
__global__ void vn_kernel(const float* __restrict__ Vv, const _Float16* __restrict__ s16,
                          _Float16* __restrict__ cat16) {
  int i = blockIdx.x * blockDim.x + threadIdx.x;
  if (i >= NNODE * Ff) return;
  int n = i >> 6, f = i & 63;
  float x = Vv[((size_t)n * 3 + 0) * Ff + f];
  float y = Vv[((size_t)n * 3 + 1) * Ff + f];
  float z = Vv[((size_t)n * 3 + 2) * Ff + f];
  float vn = sqrtf(x * x + y * y + z * z);
  cat16[(size_t)n * 128 + f] = s16[i];
  cat16[(size_t)n * 128 + 64 + f] = (_Float16)vn;
}

// a = silu(cat @ W1 + b1) @ W2 + b2  -> [N,192]
__global__ __launch_bounds__(32) void upd_mlp_kernel(
    const _Float16* __restrict__ cat16, const _Float16* __restrict__ B1,
    const float* __restrict__ b1, const _Float16* __restrict__ B2,
    const float* __restrict__ b2, float* __restrict__ a_out) {
  __shared__ _Float16 Hs[16 * 64];
  int m0 = blockIdx.x * 16;
  int lane = threadIdx.x & 31;
  int h = lane >> 4, nl = lane & 15;
  v16h a0 = load_a16(cat16, 128, m0, 0);
  v16h a1 = load_a16(cat16, 128, m0, 1);
  v16h a2 = load_a16(cat16, 128, m0, 2);
  v16h a3 = load_a16(cat16, 128, m0, 3);
  #pragma unroll
  for (int nt = 0; nt < 4; ++nt) {
    v8f acc = {};
    acc = wmma16(a0, load_b16(B1, 4, nt, 0), acc);
    acc = wmma16(a1, load_b16(B1, 4, nt, 1), acc);
    acc = wmma16(a2, load_b16(B1, 4, nt, 2), acc);
    acc = wmma16(a3, load_b16(B1, 4, nt, 3), acc);
    float bias = b1[nt * 16 + nl];
    #pragma unroll
    for (int j = 0; j < 8; ++j)
      Hs[(h * 8 + j) * 64 + nt * 16 + nl] = (_Float16)siluf(acc[j] + bias);
  }
  __syncthreads();
  v16h h0 = load_a16(Hs, 64, 0, 0);
  v16h h1 = load_a16(Hs, 64, 0, 1);
  #pragma unroll
  for (int nt = 0; nt < 12; ++nt) {
    v8f acc = {};
    acc = wmma16(h0, load_b16(B2, 2, nt, 0), acc);
    acc = wmma16(h1, load_b16(B2, 2, nt, 1), acc);
    float bias = b2[nt * 16 + nl];
    #pragma unroll
    for (int j = 0; j < 8; ++j)
      a_out[(size_t)(m0 + h * 8 + j) * 192 + nt * 16 + nl] = acc[j] + bias;
  }
}

// s += a_ss + a_sv * sum_c(Uv*Vv) ; v += a_vv * Uv ; refresh f16 copies
__global__ void apply_kernel(const float* __restrict__ a, const float* __restrict__ Uv,
                             const float* __restrict__ Vv, float* __restrict__ s,
                             float* __restrict__ v, _Float16* __restrict__ s16,
                             _Float16* __restrict__ v16b) {
  int i = blockIdx.x * blockDim.x + threadIdx.x;
  if (i >= NNODE * Ff) return;
  int n = i >> 6, f = i & 63;
  float ass = a[(size_t)n * 192 + f];
  float asv = a[(size_t)n * 192 + 64 + f];
  float avv = a[(size_t)n * 192 + 128 + f];
  float dot = 0.0f, u[3];
  #pragma unroll
  for (int c = 0; c < 3; ++c) {
    size_t idx = ((size_t)n * 3 + c) * Ff + f;
    u[c] = Uv[idx];
    dot += u[c] * Vv[idx];
  }
  float sn = s[i] + ass + asv * dot;
  s[i] = sn;
  s16[i] = (_Float16)sn;
  #pragma unroll
  for (int c = 0; c < 3; ++c) {
    size_t idx = ((size_t)n * 3 + c) * Ff + f;
    float vv = v[idx] + avv * u[c];
    v[idx] = vv;
    v16b[idx] = (_Float16)vv;
  }
}

// Gated equivariant output head (small, fp32 VALU, one thread per node)
__global__ void output_kernel(const float* __restrict__ s, const float* __restrict__ v,
                              const float* __restrict__ Wv1, const float* __restrict__ Wv2,
                              const float* __restrict__ W1, const float* __restrict__ b1,
                              const float* __restrict__ W2, const float* __restrict__ b2,
                              float* __restrict__ out) {
  int n = blockIdx.x * blockDim.x + threadIdx.x;
  if (n >= NNODE) return;
  float v1[3] = {0, 0, 0}, v2[3] = {0, 0, 0};
  for (int f = 0; f < Ff; ++f) {
    float w1 = Wv1[f], w2 = Wv2[f];
    #pragma unroll
    for (int c = 0; c < 3; ++c) {
      float vv = v[((size_t)n * 3 + c) * Ff + f];
      v1[c] += vv * w1;
      v2[c] += vv * w2;
    }
  }
  float nrm = sqrtf(v2[0] * v2[0] + v2[1] * v2[1] + v2[2] * v2[2]);
  float g = b2[1];
  for (int k = 0; k < Ff; ++k) {
    float acc = b1[k];
    for (int f = 0; f < Ff; ++f) acc += s[(size_t)n * Ff + f] * W1[(size_t)f * Ff + k];
    acc += nrm * W1[(size_t)Ff * Ff + k];
    g += siluf(acc) * W2[k * 2 + 1];
  }
  #pragma unroll
  for (int c = 0; c < 3; ++c) out[n * 3 + c] = v1[c] * g;
}

// ---------------------------------------------------------------------------
extern "C" void kernel_launch(void* const* d_in, const int* in_sizes, int n_in,
                              void* d_out, int out_size, void* d_ws, size_t ws_size,
                              hipStream_t stream) {
  const float* rp      = (const float*)d_in[0];
  const float* pp      = (const float*)d_in[1];
  const float* ip      = (const float*)d_in[2];
  const float* timev   = (const float*)d_in[3];
  const float* emb     = (const float*)d_in[4];
  const float* fW      = (const float*)d_in[5];
  const float* lvW     = (const float*)d_in[6];
  const float* phiW1   = (const float*)d_in[7];
  const float* phib1   = (const float*)d_in[8];
  const float* phiW2   = (const float*)d_in[9];
  const float* phib2   = (const float*)d_in[10];
  const float* filtW   = (const float*)d_in[11];
  const float* filtB   = (const float*)d_in[12];
  const float* updU    = (const float*)d_in[13];
  const float* updV    = (const float*)d_in[14];
  const float* updW1   = (const float*)d_in[15];
  const float* updb1   = (const float*)d_in[16];
  const float* updW2   = (const float*)d_in[17];
  const float* updb2   = (const float*)d_in[18];
  const float* oWv1    = (const float*)d_in[19];
  const float* oWv2    = (const float*)d_in[20];
  const float* oW1     = (const float*)d_in[21];
  const float* ob1     = (const float*)d_in[22];
  const float* oW2     = (const float*)d_in[23];
  const float* ob2     = (const float*)d_in[24];
  const int*   ei      = (const int*)d_in[25];
  const int*   Z       = (const int*)d_in[26];
  float* out = (float*)d_out;

  char* ws = (char*)d_ws;
  size_t off = 0;
  auto alloc = [&](size_t bytes) -> void* {
    off = (off + 255) & ~(size_t)255;
    void* p = ws + off;
    off += bytes;
    return p;
  };

  float*    s     = (float*)alloc((size_t)NNODE * Ff * 4);
  float*    v     = (float*)alloc((size_t)NNODE * 192 * 4);
  _Float16* s16   = (_Float16*)alloc((size_t)NNODE * Ff * 2);
  _Float16* v16b  = (_Float16*)alloc((size_t)NNODE * 192 * 2);
  float*    s_acc = (float*)alloc((size_t)NNODE * Ff * 4);
  float*    v_acc = (float*)alloc((size_t)NNODE * 192 * 4);
  float*    phi   = (float*)alloc((size_t)NNODE * 192 * 4);
  float*    Uv    = (float*)alloc((size_t)NNODE * 192 * 4);
  float*    Vv    = (float*)alloc((size_t)NNODE * 192 * 4);
  _Float16* cat16 = (_Float16*)alloc((size_t)NNODE * 128 * 2);
  float*    a_buf = (float*)alloc((size_t)NNODE * 192 * 4);
  _Float16* rb16  = (_Float16*)alloc((size_t)3 * NEDGE * 32 * 2);
  float*    unit  = (float*)alloc((size_t)3 * NEDGE * 3 * 4);
  float*    fcb   = (float*)alloc((size_t)3 * NEDGE * 4);

  // Packed f16 weights (per-layer offsets in halves)
  _Float16* Bphi1 = (_Float16*)alloc((size_t)NLAY * 4096 * 2);   // 64x64 -> [4][2][32][16]
  _Float16* Bphi2 = (_Float16*)alloc((size_t)NLAY * 12288 * 2);  // 64x192 -> [12][2]
  _Float16* Bfilt = (_Float16*)alloc((size_t)NLAY * 3 * 6144 * 2); // 20x192 -> [12][1]
  _Float16* BU    = (_Float16*)alloc((size_t)NLAY * 4096 * 2);
  _Float16* BV    = (_Float16*)alloc((size_t)NLAY * 4096 * 2);
  _Float16* BW1   = (_Float16*)alloc((size_t)NLAY * 8192 * 2);   // 128x64 -> [4][4]
  _Float16* BW2   = (_Float16*)alloc((size_t)NLAY * 12288 * 2);

  auto packB = [&](const float* src, _Float16* dst, int K, int Nc) {
    int KT = (K + 31) / 32;
    int total = (Nc / 16) * KT * 512;
    pack_b_kernel<<<(total + 255) / 256, 256, 0, stream>>>(src, dst, K, Nc);
  };

  for (int l = 0; l < NLAY; ++l) {
    packB(phiW1 + (size_t)l * 64 * 64,   Bphi1 + (size_t)l * 4096,  64, 64);
    packB(phiW2 + (size_t)l * 64 * 192,  Bphi2 + (size_t)l * 12288, 64, 192);
    for (int g = 0; g < 3; ++g)
      packB(filtW + ((size_t)l * 3 + g) * 20 * 192,
            Bfilt + ((size_t)l * 3 + g) * 6144, 20, 192);
    packB(updU + (size_t)l * 64 * 64,    BU + (size_t)l * 4096,     64, 64);
    packB(updV + (size_t)l * 64 * 64,    BV + (size_t)l * 4096,     64, 64);
    packB(updW1 + (size_t)l * 128 * 64,  BW1 + (size_t)l * 8192,    128, 64);
    packB(updW2 + (size_t)l * 64 * 192,  BW2 + (size_t)l * 12288,   64, 192);
  }

  geom_kernel<<<(3 * NEDGE + 255) / 256, 256, 0, stream>>>(rp, pp, ip, ei, rb16, unit, fcb);
  embed_kernel<<<(NNODE * Ff + 255) / 256, 256, 0, stream>>>(rp, pp, timev, emb, fW, lvW,
                                                             Z, s, v, s16, v16b);

  const int NT16 = NNODE / 16;       // 625
  const int RT16 = (NNODE * 3) / 16; // 1875
  const int ETILES = NEDGE / 16;     // 20000 -> 2500 blocks of 8 waves

  for (int l = 0; l < NLAY; ++l) {
    phi_kernel<<<NT16, 32, 0, stream>>>(s16, Bphi1 + (size_t)l * 4096,
                                        phib1 + (size_t)l * 64,
                                        Bphi2 + (size_t)l * 12288,
                                        phib2 + (size_t)l * 192, phi);
    hipMemsetAsync(s_acc, 0, (size_t)NNODE * Ff * 4, stream);
    hipMemsetAsync(v_acc, 0, (size_t)NNODE * 192 * 4, stream);
    edge_kernel<<<ETILES / 8, 256, 0, stream>>>(rb16, unit, fcb, phi, v, ei,
                                                Bfilt + (size_t)l * 3 * 6144,
                                                filtB + (size_t)l * 3 * 192,
                                                s_acc, v_acc);
    add_acc_kernel<<<(NNODE * Ff + 255) / 256, 256, 0, stream>>>(s, s_acc, v, v_acc,
                                                                 s16, v16b);
    uvv_kernel<<<RT16, 32, 0, stream>>>(v16b, BU + (size_t)l * 4096,
                                        BV + (size_t)l * 4096, Uv, Vv);
    vn_kernel<<<(NNODE * Ff + 255) / 256, 256, 0, stream>>>(Vv, s16, cat16);
    upd_mlp_kernel<<<NT16, 32, 0, stream>>>(cat16, BW1 + (size_t)l * 8192,
                                            updb1 + (size_t)l * 64,
                                            BW2 + (size_t)l * 12288,
                                            updb2 + (size_t)l * 192, a_buf);
    apply_kernel<<<(NNODE * Ff + 255) / 256, 256, 0, stream>>>(a_buf, Uv, Vv, s, v,
                                                               s16, v16b);
  }

  output_kernel<<<(NNODE + 255) / 256, 256, 0, stream>>>(s, v, oWv1, oWv2, oW1, ob1,
                                                         oW2, ob2, out);
}
